// DGAGeoGeneration_25735444037773
// MI455X (gfx1250) — matrix operand order repaired and checked by reading, not compiled, not run on hardware
//
#include <hip/hip_runtime.h>
#include <hip/hip_bf16.h>
#include <cstdint>

#define BATCH 2
#define NPTS  4096
#define CIN   128
#define DIMD  64
#define FPSN  1024

typedef __attribute__((ext_vector_type(16))) _Float16 v16h;
typedef __attribute__((ext_vector_type(8)))  _Float16 v8h;
typedef __attribute__((ext_vector_type(8)))  float    v8f;

// ---------------------------------------------------------------------------
// WMMA GEMM: OutT[n, m] = post( (W @ X)[m, n] )   (f16 out, row stride = M)
//   post(v) = relu?( v*scale[m] + shift[m] + resid[n,m] )
// Wsw: A-fragments pre-swizzled per (mtile, kstep): [tile][lane 0..31][h 0..15]
// XT : [Nn, K] f16 row-major  (B fragment: lane -> col n, 16 consecutive K)
// One wave computes a 16x16 tile; 4 waves/block share the same m-tile.
// ---------------------------------------------------------------------------
__global__ __launch_bounds__(128) void gemm_wmma_kernel(
    const _Float16* __restrict__ Wsw, const _Float16* __restrict__ XT,
    const float* __restrict__ scale, const float* __restrict__ shift,
    const _Float16* __restrict__ resid, _Float16* __restrict__ OutT,
    int M, int K, int Nn, int relu)
{
  const int lane = threadIdx.x & 31;
  const int wave = threadIdx.x >> 5;
  const int nt = blockIdx.x * 4 + wave;
  const int mt = blockIdx.y;
  if (nt * 16 >= Nn) return;                 // whole-wave uniform: EXEC stays all-1s
  const int ksteps = K >> 5;
  const int n  = nt * 16 + (lane & 15);
  const int hi = lane >> 4;                  // which K-half this half-wave owns
  v8f acc = {};
  const _Float16* wp = Wsw + ((size_t)mt * ksteps) * 512 + lane * 16;
  const _Float16* xp = XT + (size_t)n * K + hi * 16;
  for (int ks = 0; ks < ksteps; ++ks) {
    v16h a = *(const v16h*)wp;  wp += 512;
    v16h b = *(const v16h*)xp;  xp += 32;
    acc = __builtin_amdgcn_wmma_f32_16x16x32_f16(false, a, false, b, (short)0, acc,
                                                 false, false);
  }
  const int mbase = mt * 16 + hi * 8;        // C/D layout: VGPR r -> row mbase+r
  v8h o;
#pragma unroll
  for (int r = 0; r < 8; ++r) {
    int m = mbase + r;
    float v = acc[r] * scale[m] + shift[m];
    if (resid) v += (float)resid[(size_t)n * M + m];
    if (relu)  v = v > 0.0f ? v : 0.0f;
    o[r] = (_Float16)v;
  }
  *(v8h*)(OutT + (size_t)n * M + mbase) = o;
}

// Swizzle W[M,K] f32 -> A-fragment layout f16 (lane<16: rows, contiguous K = 16*(lane>>4) + h)
__global__ void swizzle_w_kernel(const float* __restrict__ W, _Float16* __restrict__ Wsw,
                                 int M, int K)
{
  int idx = blockIdx.x * blockDim.x + threadIdx.x;
  if (idx >= M * K) return;
  int h    = idx & 15;
  int lane = (idx >> 4) & 31;
  int t    = idx >> 9;            // tile = mt*ksteps + ks
  int ksteps = K >> 5;
  int ks = t % ksteps;
  int mt = t / ksteps;
  int row = mt * 16 + (lane & 15);
  int k   = ks * 32 + ((lane >> 4) << 4) + h;
  Wsw[idx] = (_Float16)W[(size_t)row * K + k];
}

// Fold bias (+ optional eval-mode BN) into per-channel scale/shift
__global__ void scale_shift_kernel(const float* __restrict__ bias, const float* __restrict__ g,
                                   const float* __restrict__ beta, const float* __restrict__ mean,
                                   const float* __restrict__ var,
                                   float* __restrict__ scale, float* __restrict__ shift, int M)
{
  int m = blockIdx.x * blockDim.x + threadIdx.x;
  if (m >= M) return;
  if (g) {
    float inv = g[m] * rsqrtf(var[m] + 1e-5f);
    scale[m] = inv;
    shift[m] = bias[m] * inv + beta[m] - mean[m] * inv;
  } else {
    scale[m] = 1.0f;
    shift[m] = bias[m];
  }
}

// [B,3,N] -> [B,N,3]
__global__ void point_major_kernel(const float* __restrict__ P, float* __restrict__ Pp, int total)
{
  int i = blockIdx.x * blockDim.x + threadIdx.x;
  if (i >= total) return;
  int b = i / NPTS, n = i % NPTS;
#pragma unroll
  for (int c = 0; c < 3; ++c)
    Pp[(size_t)i * 3 + c] = P[((size_t)b * 3 + c) * NPTS + n];
}

// [B,C,N] f32 -> [(b,n), C] f16
__global__ void feat_T16_kernel(const float* __restrict__ F, _Float16* __restrict__ FT, int total)
{
  int t = blockIdx.x * blockDim.x + threadIdx.x;
  if (t >= total) return;
  int row = t / CIN, c = t % CIN;
  int b = row / NPTS, n = row % NPTS;
  FT[t] = (_Float16)F[((size_t)b * CIN + c) * NPTS + n];
}

// Furthest point sampling: one block per batch, LDS-resident running distances.
__global__ __launch_bounds__(1024) void fps_kernel(const float* __restrict__ pqP,
                                                   int* __restrict__ fpsIdx)
{
  __shared__ float dists[NPTS];
  __shared__ float rv[1024];
  __shared__ int   ri[1024];
  __shared__ int   sfar;
  const int b = blockIdx.x, tid = threadIdx.x;
  const float* base = pqP + (size_t)b * NPTS * 3;
  for (int j = tid; j < NPTS; j += 1024) dists[j] = 1e10f;
  if (tid == 0) sfar = 0;
  __syncthreads();
  for (int s = 0; s < FPSN; ++s) {
    const int far = sfar;
    if (tid == 0) fpsIdx[b * FPSN + s] = far;
    const float cx = base[far * 3], cy = base[far * 3 + 1], cz = base[far * 3 + 2];
    float bv = -1.0f; int bi = 0;
    for (int j = tid; j < NPTS; j += 1024) {
      float dx = base[j * 3] - cx, dy = base[j * 3 + 1] - cy, dz = base[j * 3 + 2] - cz;
      float d = dx * dx + dy * dy + dz * dz;
      float dm = fminf(dists[j], d);
      dists[j] = dm;
      if (dm > bv) { bv = dm; bi = j; }
    }
    rv[tid] = bv; ri[tid] = bi;
    __syncthreads();
    for (int o = 512; o > 0; o >>= 1) {
      if (tid < o) {
        float ov = rv[tid + o]; int oi = ri[tid + o];
        if (ov > rv[tid] || (ov == rv[tid] && oi < ri[tid])) { rv[tid] = ov; ri[tid] = oi; }
      }
      __syncthreads();
    }
    if (tid == 0) sfar = ri[0];
    __syncthreads();
  }
}

// gather positions by FPS index (fidx==null -> identity)
__global__ void gather_pos_kernel(const float* __restrict__ pqP, const int* __restrict__ fidx,
                                  float* __restrict__ dst, int Nq, int total)
{
  int i = blockIdx.x * blockDim.x + threadIdx.x;
  if (i >= total) return;
  int b = i / Nq, n = i % Nq;
  int s = fidx ? fidx[b * FPSN + n] : n;
#pragma unroll
  for (int c = 0; c < 3; ++c)
    dst[(size_t)i * 3 + c] = pqP[((size_t)b * NPTS + s) * 3 + c];
}

// gather feature rows (f16) with destination stride/offset (for concat)
__global__ void gather_feat_kernel(const _Float16* __restrict__ src, const int* __restrict__ fidx,
                                   _Float16* __restrict__ dst, int Nq, int dstStride, int dstOff,
                                   int total)
{
  int t = blockIdx.x * blockDim.x + threadIdx.x;
  if (t >= total) return;
  int row = t / CIN, c = t % CIN;
  int b = row / Nq, n = row % Nq;
  int s = fidx ? fidx[b * FPSN + n] : n;
  dst[(size_t)row * dstStride + dstOff + c] = src[((size_t)b * NPTS + s) * CIN + c];
}

// brute-force KNN of queries vs the (LDS-cached) 4096-point source cloud
__global__ __launch_bounds__(256) void knn_kernel(const float* __restrict__ pos1P,
                                                  const float* __restrict__ psP,
                                                  int* __restrict__ knnidx, int Nq, int k)
{
  __shared__ float sx[NPTS], sy[NPTS], sz[NPTS];
  const int b = blockIdx.y;
  for (int j = threadIdx.x; j < NPTS; j += 256) {
    sx[j] = psP[((size_t)b * NPTS + j) * 3 + 0];
    sy[j] = psP[((size_t)b * NPTS + j) * 3 + 1];
    sz[j] = psP[((size_t)b * NPTS + j) * 3 + 2];
  }
  __syncthreads();
  int n = blockIdx.x * 256 + threadIdx.x;
  if (n >= Nq) return;
  const float qx = pos1P[((size_t)b * Nq + n) * 3 + 0];
  const float qy = pos1P[((size_t)b * Nq + n) * 3 + 1];
  const float qz = pos1P[((size_t)b * Nq + n) * 3 + 2];
  float kd[16]; int ki[16];
#pragma unroll
  for (int t = 0; t < 16; ++t) { kd[t] = 3.4e38f; ki[t] = 0; }
  for (int j = 0; j < NPTS; ++j) {
    float dx = qx - sx[j], dy = qy - sy[j], dz = qz - sz[j];
    float d = dx * dx + dy * dy + dz * dz;
#pragma unroll
    for (int t = 15; t >= 0; --t) {
      float pv = (t > 0) ? kd[t - 1] : -1.0f;
      int   pi = (t > 0) ? ki[t - 1] : 0;
      if (d < kd[t]) {
        if (t == 0 || pv <= d) { kd[t] = d; ki[t] = j; }
        else                   { kd[t] = pv; ki[t] = pi; }
      }
    }
  }
  for (int t = 0; t < k; ++t) knnidx[((size_t)b * Nq + n) * k + t] = ki[t];
}

// 3-NN + inverse-distance weights vs previous-level points (<=1024, LDS cached)
__global__ __launch_bounds__(256) void three_nn_kernel(const float* __restrict__ tgtP,
                                                       const float* __restrict__ srcP,
                                                       int* __restrict__ idx3, float* __restrict__ w3,
                                                       int Nq, int Mprev)
{
  __shared__ float sx[1024], sy[1024], sz[1024];
  const int b = blockIdx.y;
  for (int j = threadIdx.x; j < Mprev; j += 256) {
    sx[j] = srcP[((size_t)b * Mprev + j) * 3 + 0];
    sy[j] = srcP[((size_t)b * Mprev + j) * 3 + 1];
    sz[j] = srcP[((size_t)b * Mprev + j) * 3 + 2];
  }
  __syncthreads();
  int n = blockIdx.x * 256 + threadIdx.x;
  if (n >= Nq) return;
  const float qx = tgtP[((size_t)b * Nq + n) * 3 + 0];
  const float qy = tgtP[((size_t)b * Nq + n) * 3 + 1];
  const float qz = tgtP[((size_t)b * Nq + n) * 3 + 2];
  float kd[3] = {3.4e38f, 3.4e38f, 3.4e38f};
  int   ki[3] = {0, 0, 0};
  for (int j = 0; j < Mprev; ++j) {
    float dx = qx - sx[j], dy = qy - sy[j], dz = qz - sz[j];
    float d = dx * dx + dy * dy + dz * dz;
#pragma unroll
    for (int t = 2; t >= 0; --t) {
      float pv = (t > 0) ? kd[t - 1] : -1.0f;
      int   pi = (t > 0) ? ki[t - 1] : 0;
      if (d < kd[t]) {
        if (t == 0 || pv <= d) { kd[t] = d; ki[t] = j; }
        else                   { kd[t] = pv; ki[t] = pi; }
      }
    }
  }
  float r0 = 1.0f / fmaxf(kd[0], 1e-10f);
  float r1 = 1.0f / fmaxf(kd[1], 1e-10f);
  float r2 = 1.0f / fmaxf(kd[2], 1e-10f);
  float inv = 1.0f / (r0 + r1 + r2);
  size_t o = ((size_t)b * Nq + n) * 3;
  idx3[o + 0] = ki[0]; idx3[o + 1] = ki[1]; idx3[o + 2] = ki[2];
  w3[o + 0] = r0 * inv; w3[o + 1] = r1 * inv; w3[o + 2] = r2 * inv;
}

// 3-point weighted interpolation of previous-level features into concat buffer
__global__ void interp_kernel(const _Float16* __restrict__ preF, const int* __restrict__ idx3,
                              const float* __restrict__ w3, _Float16* __restrict__ dst,
                              int Nq, int Mprev, int dstStride, int dstOff, int total)
{
  int t = blockIdx.x * blockDim.x + threadIdx.x;
  if (t >= total) return;
  int row = t / CIN, c = t % CIN;
  int b = row / Nq;
  float s = 0.0f;
#pragma unroll
  for (int j = 0; j < 3; ++j)
    s += w3[(size_t)row * 3 + j] *
         (float)preF[((size_t)b * Mprev + idx3[(size_t)row * 3 + j]) * CIN + c];
  dst[(size_t)row * dstStride + dstOff + c] = (_Float16)s;
}

// pe1 = relu(bn(pw1 @ pos_rel + pb1))   (K=3 -> scalar, BN folded into scale/shift)
__global__ void pe1_kernel(const float* __restrict__ pos1P, const float* __restrict__ psP,
                           const int* __restrict__ knnidx, const float* __restrict__ pw1,
                           const float* __restrict__ scale, const float* __restrict__ shift,
                           _Float16* __restrict__ pe1T, int Nq, int k, int total)
{
  int t = blockIdx.x * blockDim.x + threadIdx.x;
  if (t >= total) return;
  int f = t >> 6, m = t & 63;
  int bn = f / k;
  int b = bn / Nq;
  int j = knnidx[f];
  float rx = pos1P[(size_t)bn * 3 + 0] - psP[((size_t)b * NPTS + j) * 3 + 0];
  float ry = pos1P[(size_t)bn * 3 + 1] - psP[((size_t)b * NPTS + j) * 3 + 1];
  float rz = pos1P[(size_t)bn * 3 + 2] - psP[((size_t)b * NPTS + j) * 3 + 2];
  float a = pw1[m * 3 + 0] * rx + pw1[m * 3 + 1] * ry + pw1[m * 3 + 2] * rz;
  a = a * scale[m] + shift[m];
  pe1T[t] = (_Float16)(a > 0.0f ? a : 0.0f);
}

// x = q - key[knn] + pe ;  v = val[knn] + pe
__global__ void build_xv_kernel(const _Float16* __restrict__ qT, const _Float16* __restrict__ keyT,
                                const _Float16* __restrict__ valT, const _Float16* __restrict__ peT,
                                const int* __restrict__ knnidx, _Float16* __restrict__ xT,
                                _Float16* __restrict__ vT, int Nq, int k, int total)
{
  int t = blockIdx.x * blockDim.x + threadIdx.x;
  if (t >= total) return;
  int f = t >> 6, c = t & 63;
  int bn = f / k;
  int b = bn / Nq;
  int j = knnidx[f];
  float pe = (float)peT[t];
  float q  = (float)qT[(size_t)bn * 64 + c];
  float ke = (float)keyT[((size_t)b * NPTS + j) * 64 + c];
  float va = (float)valT[((size_t)b * NPTS + j) * 64 + c];
  xT[t] = (_Float16)(q - ke + pe);
  vT[t] = (_Float16)(va + pe);
}

// per (point, channel) softmax over k neighbors, then weighted aggregation
__global__ void softmax_agg_kernel(const _Float16* __restrict__ aT, const _Float16* __restrict__ vT,
                                   _Float16* __restrict__ aggT, int k, int total)
{
  int t = blockIdx.x * blockDim.x + threadIdx.x;
  if (t >= total) return;
  int bn = t >> 6, c = t & 63;
  size_t base = (size_t)bn * k * 64 + c;
  float mx = -3.4e38f;
  for (int kk = 0; kk < k; ++kk) mx = fmaxf(mx, (float)aT[base + (size_t)kk * 64]);
  float s = 0.0f, acc = 0.0f;
  for (int kk = 0; kk < k; ++kk) {
    float e = __expf((float)aT[base + (size_t)kk * 64] - mx);
    s += e;
    acc += e * (float)vT[base + (size_t)kk * 64];
  }
  aggT[t] = (_Float16)(acc / s);
}

// [(b,n),C] f16 -> [B,C,N] f32 output
__global__ void final_out_kernel(const _Float16* __restrict__ fT, float* __restrict__ out, int total)
{
  int t = blockIdx.x * blockDim.x + threadIdx.x;
  if (t >= total) return;
  int b = t / (CIN * NPTS);
  int rem = t % (CIN * NPTS);
  int c = rem / NPTS, n = rem % NPTS;
  out[t] = (float)fT[((size_t)b * NPTS + n) * CIN + c];
}

// ---------------------------------------------------------------------------
// Host orchestration
// ---------------------------------------------------------------------------
struct DgaP {
  const float *ab1, *ab2, *abn_g, *abn_b, *abn_m, *abn_v, *aw1, *aw2, *be, *bk, *bq, *bv,
              *pb1, *pb2, *pbn_g, *pbn_b, *pbn_m, *pbn_v, *pw1, *pw2, *we, *wk, *wq, *wv;
};
struct QmlpP { const float *b1, *b2, *bs, *w1, *w2, *ws; };

extern "C" void kernel_launch(void* const* d_in, const int* in_sizes, int n_in,
                              void* d_out, int out_size, void* d_ws, size_t ws_size,
                              hipStream_t stream)
{
  (void)in_sizes; (void)n_in; (void)out_size; (void)ws_size;
  const float* pq = (const float*)d_in[0];
  const float* fq = (const float*)d_in[1];
  const float* ps = (const float*)d_in[2];
  const float* fs = (const float*)d_in[3];
  int pi = 4;
  DgaP dga[3];
  for (int i = 0; i < 3; ++i) {
    const float** q = (const float**)&dga[i];
    for (int j = 0; j < 24; ++j) q[j] = (const float*)d_in[pi++];
  }
  QmlpP qm[2];
  for (int i = 0; i < 2; ++i) {
    const float** q = (const float**)&qm[i];
    for (int j = 0; j < 6; ++j) q[j] = (const float*)d_in[pi++];
  }

  size_t off = 0;
  auto alloc = [&](size_t bytes) -> void* {
    void* p = (char*)d_ws + off;
    off = (off + bytes + 255) & ~(size_t)255;
    return p;
  };

  const size_t RP = (size_t)BATCH * NPTS;         // max rows
  const size_t Fmax = RP * 16;                    // max flattened (point,neighbor)
  float*    pqP   = (float*)alloc(RP * 3 * 4);
  float*    psP   = (float*)alloc(RP * 3 * 4);
  float*    posA  = (float*)alloc(RP * 3 * 4);
  float*    posB  = (float*)alloc(RP * 3 * 4);
  _Float16* fqT   = (_Float16*)alloc(RP * CIN * 2);
  _Float16* fsT   = (_Float16*)alloc(RP * CIN * 2);
  _Float16* keyT  = (_Float16*)alloc(RP * DIMD * 2);
  _Float16* valT  = (_Float16*)alloc(RP * DIMD * 2);
  _Float16* fA    = (_Float16*)alloc(RP * CIN * 2);
  _Float16* fB    = (_Float16*)alloc(RP * CIN * 2);
  _Float16* f1T   = (_Float16*)alloc(RP * CIN * 2);
  _Float16* catT  = (_Float16*)alloc(RP * 2 * CIN * 2);
  _Float16* scT   = (_Float16*)alloc(RP * CIN * 2);
  _Float16* mhT   = (_Float16*)alloc(RP * CIN * 2);
  _Float16* qT    = (_Float16*)alloc(RP * DIMD * 2);
  _Float16* aggT  = (_Float16*)alloc(RP * DIMD * 2);
  int*      fpsIdx = (int*)alloc((size_t)BATCH * FPSN * 4);
  int*      knnidx = (int*)alloc(RP * 16 * 4);
  int*      idx3   = (int*)alloc(RP * 3 * 4);
  float*    w3     = (float*)alloc(RP * 3 * 4);
  float*    scaleBuf = (float*)alloc(16 * 256 * 4);
  float*    shiftBuf = (float*)alloc(16 * 256 * 4);
  _Float16* WQ  = (_Float16*)alloc((size_t)DIMD * CIN * 2);
  _Float16* WK  = (_Float16*)alloc((size_t)DIMD * CIN * 2);
  _Float16* WV  = (_Float16*)alloc((size_t)DIMD * CIN * 2);
  _Float16* PW2 = (_Float16*)alloc((size_t)DIMD * DIMD * 2);
  _Float16* AW1 = (_Float16*)alloc((size_t)4 * DIMD * DIMD * 2);
  _Float16* AW2 = (_Float16*)alloc((size_t)4 * DIMD * DIMD * 2);
  _Float16* WE  = (_Float16*)alloc((size_t)CIN * DIMD * 2);
  _Float16* W1  = (_Float16*)alloc((size_t)CIN * 2 * CIN * 2);
  _Float16* W2  = (_Float16*)alloc((size_t)CIN * CIN * 2);
  _Float16* WS  = (_Float16*)alloc((size_t)CIN * 2 * CIN * 2);
  _Float16* pe1x = (_Float16*)alloc(Fmax * DIMD * 2);   // pe1, later reused as x
  _Float16* pea  = (_Float16*)alloc(Fmax * DIMD * 2);   // pe, later reused as a
  _Float16* vTb  = (_Float16*)alloc(Fmax * DIMD * 2);
  _Float16* hT   = (_Float16*)alloc(Fmax * 4 * DIMD * 2);

  auto grid1 = [](int total) { return dim3((unsigned)((total + 255) / 256)); };
  auto swz = [&](const float* W, _Float16* dst, int M, int K) {
    swizzle_w_kernel<<<grid1(M * K), 256, 0, stream>>>(W, dst, M, K);
  };
  auto ss = [&](const float* bias, const float* g, const float* bet, const float* me,
                const float* va, int slot, int M) {
    scale_shift_kernel<<<dim3(1), 256, 0, stream>>>(bias, g, bet, me, va,
                                                    scaleBuf + slot * 256,
                                                    shiftBuf + slot * 256, M);
  };
  auto gemm = [&](const _Float16* W, const _Float16* X, int slot, const _Float16* resid,
                  _Float16* out, int M, int K, int Nn, int relu) {
    gemm_wmma_kernel<<<dim3((unsigned)(Nn / 64), (unsigned)(M / 16)), 128, 0, stream>>>(
        W, X, scaleBuf + slot * 256, shiftBuf + slot * 256, resid, out, M, K, Nn, relu);
  };

  // ---- stage 0: layout conversion + FPS (prefix property: fps(512) == fps(1024)[:512])
  point_major_kernel<<<grid1(BATCH * NPTS), 256, 0, stream>>>(pq, pqP, BATCH * NPTS);
  point_major_kernel<<<grid1(BATCH * NPTS), 256, 0, stream>>>(ps, psP, BATCH * NPTS);
  feat_T16_kernel<<<grid1(BATCH * NPTS * CIN), 256, 0, stream>>>(fq, fqT, BATCH * NPTS * CIN);
  feat_T16_kernel<<<grid1(BATCH * NPTS * CIN), 256, 0, stream>>>(fs, fsT, BATCH * NPTS * CIN);
  fps_kernel<<<dim3(BATCH), 1024, 0, stream>>>(pqP, fpsIdx);

  const int NqArr[3] = {NPTS, NPTS / 4, NPTS / 8};
  const int kArr[3]  = {16, 12, 8};
  float* posBufs[2] = {posA, posB};
  _Float16* fBufs[2] = {fA, fB};
  float* posPrev = nullptr;
  _Float16* fPrev = nullptr;
  int NqPrev = 0, parity = 0;

  for (int i = 2; i >= 0; --i) {
    const int Nq = NqArr[i], k = kArr[i];
    const int rows = BATCH * Nq;
    const int F = rows * k;
    const DgaP& P = dga[i];
    const int* fidx = (i == 0) ? nullptr : fpsIdx;
    float* posCur = posBufs[parity];
    _Float16* fCur = fBufs[parity];
    parity ^= 1;

    // swizzle this level's weights + fold bias/BN epilogues
    swz(P.wq, WQ, DIMD, CIN);  swz(P.wk, WK, DIMD, CIN);  swz(P.wv, WV, DIMD, CIN);
    swz(P.pw2, PW2, DIMD, DIMD);
    swz(P.aw1, AW1, 4 * DIMD, DIMD);  swz(P.aw2, AW2, DIMD, 4 * DIMD);
    swz(P.we, WE, CIN, DIMD);
    ss(P.bq, nullptr, nullptr, nullptr, nullptr, 0, DIMD);
    ss(P.bk, nullptr, nullptr, nullptr, nullptr, 1, DIMD);
    ss(P.bv, nullptr, nullptr, nullptr, nullptr, 2, DIMD);
    ss(P.pb1, P.pbn_g, P.pbn_b, P.pbn_m, P.pbn_v, 3, DIMD);
    ss(P.pb2, nullptr, nullptr, nullptr, nullptr, 4, DIMD);
    ss(P.ab1, P.abn_g, P.abn_b, P.abn_m, P.abn_v, 5, 4 * DIMD);
    ss(P.ab2, nullptr, nullptr, nullptr, nullptr, 6, DIMD);
    ss(P.be, nullptr, nullptr, nullptr, nullptr, 7, CIN);

    gather_pos_kernel<<<grid1(rows), 256, 0, stream>>>(pqP, fidx, posCur, Nq, rows);

    if (i == 2) {
      gather_feat_kernel<<<grid1(rows * CIN), 256, 0, stream>>>(fqT, fidx, f1T, Nq, CIN, 0,
                                                                rows * CIN);
    } else {
      // concat[f1 | 3-NN interpolation of previous level], then residual MLP
      gather_feat_kernel<<<grid1(rows * CIN), 256, 0, stream>>>(fqT, fidx, catT, Nq, 2 * CIN, 0,
                                                                rows * CIN);
      three_nn_kernel<<<dim3((unsigned)((Nq + 255) / 256), BATCH), 256, 0, stream>>>(
          posCur, posPrev, idx3, w3, Nq, NqPrev);
      interp_kernel<<<grid1(rows * CIN), 256, 0, stream>>>(fPrev, idx3, w3, catT, Nq, NqPrev,
                                                           2 * CIN, CIN, rows * CIN);
      swz(qm[i].ws, WS, CIN, 2 * CIN);
      swz(qm[i].w1, W1, CIN, 2 * CIN);
      swz(qm[i].w2, W2, CIN, CIN);
      ss(qm[i].b1, nullptr, nullptr, nullptr, nullptr, 8, CIN);
      ss(qm[i].b2, nullptr, nullptr, nullptr, nullptr, 9, CIN);
      ss(qm[i].bs, nullptr, nullptr, nullptr, nullptr, 10, CIN);
      gemm(WS, catT, 10, nullptr, scT, CIN, 2 * CIN, rows, 0);
      gemm(W1, catT, 8, nullptr, mhT, CIN, 2 * CIN, rows, 1);
      gemm(W2, mhT, 9, scT, f1T, CIN, CIN, rows, 0);
    }

    // DGA attention
    gemm(WK, fsT, 1, nullptr, keyT, DIMD, CIN, BATCH * NPTS, 0);
    gemm(WV, fsT, 2, nullptr, valT, DIMD, CIN, BATCH * NPTS, 0);
    gemm(WQ, f1T, 0, nullptr, qT, DIMD, CIN, rows, 0);
    knn_kernel<<<dim3((unsigned)((Nq + 255) / 256), BATCH), 256, 0, stream>>>(posCur, psP,
                                                                              knnidx, Nq, k);
    pe1_kernel<<<grid1(F * DIMD), 256, 0, stream>>>(posCur, psP, knnidx, P.pw1,
                                                    scaleBuf + 3 * 256, shiftBuf + 3 * 256,
                                                    pe1x, Nq, k, F * DIMD);
    gemm(PW2, pe1x, 4, nullptr, pea, DIMD, DIMD, F, 0);             // pe
    build_xv_kernel<<<grid1(F * DIMD), 256, 0, stream>>>(qT, keyT, valT, pea, knnidx,
                                                         pe1x /*x*/, vTb, Nq, k, F * DIMD);
    gemm(AW1, pe1x, 5, nullptr, hT, 4 * DIMD, DIMD, F, 1);          // hidden (BN+ReLU)
    gemm(AW2, hT, 6, nullptr, pea /*a*/, DIMD, 4 * DIMD, F, 0);     // logits
    softmax_agg_kernel<<<grid1(rows * DIMD), 256, 0, stream>>>(pea, vTb, aggT, k, rows * DIMD);
    gemm(WE, aggT, 7, f1T, fCur, CIN, DIMD, rows, 0);               // + residual f1

    posPrev = posCur; fPrev = fCur; NqPrev = Nq;
  }

  final_out_kernel<<<grid1(BATCH * CIN * NPTS), 256, 0, stream>>>(fPrev, (float*)d_out,
                                                                  BATCH * CIN * NPTS);
}
